// GCNEncoderWithPool_4904852652785
// MI455X (gfx1250) — matrix operand order repaired and checked by reading, not compiled, not run on hardware
//
#include <hip/hip_runtime.h>
#include <hip/hip_bf16.h>
#include <math.h>

typedef __attribute__((ext_vector_type(16))) __bf16 v16bf;
typedef __attribute__((ext_vector_type(8)))  float  v8f;

#define NN   65536
#define EE   1048576
#define IND  192
#define PED  64
#define CIN  256
#define HID  256
#define LAT  128
#define NGR  64
#define NPER 1024
#define KPER 512
#define POOL (NGR * KPER)

__device__ __forceinline__ size_t gtid() {
  return (size_t)blockIdx.x * blockDim.x + threadIdx.x;
}
#define GSTRIDE ((size_t)gridDim.x * blockDim.x)

// ---------------- utility ----------------
__global__ void k_fill(float* __restrict__ p, float v, size_t n) {
  for (size_t t = gtid(); t < n; t += GSTRIDE) p[t] = v;
}

__global__ void k_deg(const int* __restrict__ dst, float* __restrict__ deg) {
  for (size_t t = gtid(); t < (size_t)EE; t += GSTRIDE)
    atomicAdd(&deg[dst[t]], 1.0f);
}

__global__ void k_rsqrt(float* __restrict__ d) {
  size_t t = gtid();
  if (t < (size_t)NN) d[t] = rsqrtf(d[t]);
}

// ============ WMMA GEMM with LDS-staged A (bf16, fragment order) ============
// Block = 256 threads = 8 waves. Block stages one 16-row x 256-K strip of A
// into LDS in WMMA-fragment order, then each wave computes one 16x16 output
// tile per column (8 column tiles per block) with 8 chained
// v_wmma_f32_16x16x32_bf16 ops, A fed by ds_load_b128 pairs.

// A-element fetch for GEMM1 (concat of x_raw[.,192] and pe[.,64])
__device__ __forceinline__ float fetchA1(const float* xraw, const float* pe,
                                         int m, int k) {
  return (k < IND) ? xraw[(size_t)m * IND + k] : pe[(size_t)m * PED + (k - IND)];
}

__global__ void __launch_bounds__(256)
k_gemm1(const float* __restrict__ xraw, const float* __restrict__ pe,
        const float* __restrict__ W1, float* __restrict__ XW) {
  __shared__ v16bf As[256];                 // 8 kb-blocks * 32 lanes, 8 KB
  __bf16* Asb = (__bf16*)As;

  int rowTile = blockIdx.x >> 1;            // 4096 row tiles
  int colBase = (blockIdx.x & 1) * 8;       // 2 column groups of 8 tiles

  // ---- stage A strip: thread t fills fragment slot (kb = t>>5, lane = t&31)
  {
    int t   = threadIdx.x;
    int kb  = t >> 5;
    int ln  = t & 31;
    int m   = rowTile * 16 + (ln & 15);
    int hf  = ln >> 4;
#pragma unroll
    for (int i = 0; i < 8; ++i) {
      int k = kb * 32 + ((i & 4) ? 16 : 0) + hf * 8 + 2 * (i & 3);
      Asb[(size_t)t * 16 + 2 * i]     = (__bf16)fetchA1(xraw, pe, m, k);
      Asb[(size_t)t * 16 + 2 * i + 1] = (__bf16)fetchA1(xraw, pe, m, k + 1);
    }
  }
  __syncthreads();

  // ---- compute: wave w -> column tile colBase + w
  int w    = threadIdx.x >> 5;
  int lane = threadIdx.x & 31;
  int colTile = colBase + w;
  int hf = lane >> 4;
  v8f c = {};
#pragma unroll
  for (int kb = 0; kb < CIN / 32; ++kb) {
    v16bf a = As[kb * 32 + lane];           // ds_load_b128 x2, conflict-free
    v16bf b;
    const float* wp = W1 + (size_t)(kb * 32 + hf * 8) * HID + colTile * 16 + (lane & 15);
#pragma unroll
    for (int i = 0; i < 8; ++i) {
      b[2 * i]     = (__bf16)wp[(size_t)(((i & 4) ? 16 : 0) + 2 * (i & 3)) * HID];
      b[2 * i + 1] = (__bf16)wp[(size_t)(((i & 4) ? 16 : 0) + 2 * (i & 3) + 1) * HID];
    }
    c = __builtin_amdgcn_wmma_f32_16x16x32_bf16(false, a, false, b, (short)0, c,
                                                false, false);
  }
#pragma unroll
  for (int r = 0; r < 8; ++r) {
    int mr = rowTile * 16 + hf * 8 + r;
    XW[(size_t)mr * HID + colTile * 16 + (lane & 15)] = c[r];
  }
}

// generic: A[NN x 256] fp32 @ W[256 x OUTD] -> Out ; OUTD = 128 here (8 tiles)
__global__ void __launch_bounds__(256)
k_gemm_h(const float* __restrict__ A, const float* __restrict__ W,
         float* __restrict__ Out, int OUTD) {
  __shared__ v16bf As[256];
  __bf16* Asb = (__bf16*)As;

  int rowTile = blockIdx.x;                 // 4096 row tiles, 8 col tiles/block
  {
    int t  = threadIdx.x;
    int kb = t >> 5;
    int ln = t & 31;
    const float* ap = A + (size_t)(rowTile * 16 + (ln & 15)) * CIN;
    int hf = ln >> 4;
#pragma unroll
    for (int i = 0; i < 8; ++i) {
      int k = kb * 32 + ((i & 4) ? 16 : 0) + hf * 8 + 2 * (i & 3);
      Asb[(size_t)t * 16 + 2 * i]     = (__bf16)ap[k];
      Asb[(size_t)t * 16 + 2 * i + 1] = (__bf16)ap[k + 1];
    }
  }
  __syncthreads();

  int w    = threadIdx.x >> 5;
  int lane = threadIdx.x & 31;
  int colTile = w;                          // OUTD/16 == 8 == waves per block
  int hf = lane >> 4;
  v8f c = {};
#pragma unroll
  for (int kb = 0; kb < CIN / 32; ++kb) {
    v16bf a = As[kb * 32 + lane];
    v16bf b;
    const float* wp = W + (size_t)(kb * 32 + hf * 8) * OUTD + colTile * 16 + (lane & 15);
#pragma unroll
    for (int i = 0; i < 8; ++i) {
      b[2 * i]     = (__bf16)wp[(size_t)(((i & 4) ? 16 : 0) + 2 * (i & 3)) * OUTD];
      b[2 * i + 1] = (__bf16)wp[(size_t)(((i & 4) ? 16 : 0) + 2 * (i & 3) + 1) * OUTD];
    }
    c = __builtin_amdgcn_wmma_f32_16x16x32_bf16(false, a, false, b, (short)0, c,
                                                false, false);
  }
#pragma unroll
  for (int r = 0; r < 8; ++r) {
    int mr = rowTile * 16 + hf * 8 + r;
    Out[(size_t)mr * OUTD + colTile * 16 + (lane & 15)] = c[r];
  }
}

// ---------------- edge scatter: agg[dst] += dinv[s]*dinv[d] * hin[src] ----------------
__global__ void k_prop(const int* __restrict__ src, const int* __restrict__ dst,
                       const float* __restrict__ dinv, const float* __restrict__ hin,
                       float* __restrict__ agg, int C) {
  int cq = C >> 2;
  size_t total = (size_t)EE * (size_t)cq;
  for (size_t t = gtid(); t < total; t += GSTRIDE) {
    size_t e = t / cq;
    int c = (int)(t % cq) * 4;
    int s = src[e], d = dst[e];
    float nm = dinv[s] * dinv[d];
    const float4 v = *reinterpret_cast<const float4*>(hin + (size_t)s * C + c);
    float* o = agg + (size_t)d * C + c;
    atomicAdd(o + 0, nm * v.x);
    atomicAdd(o + 1, nm * v.y);
    atomicAdd(o + 2, nm * v.z);
    atomicAdd(o + 3, nm * v.w);
  }
}

// self-loop term + bias (+ optional relu), in place on agg
__global__ void k_finish(float* __restrict__ agg, const float* __restrict__ own,
                         const float* __restrict__ dinv, const float* __restrict__ bias,
                         int C, int relu) {
  size_t total = (size_t)NN * C;
  for (size_t t = gtid(); t < total; t += GSTRIDE) {
    size_t i = t / C;
    int c = (int)(t % C);
    float di = dinv[i];
    float v = agg[t] + di * di * own[t] + bias[c];
    agg[t] = relu ? fmaxf(v, 0.0f) : v;
  }
}

// ---------------- reparameterization, score, top-k, gather ----------------
__global__ void k_z(const float* __restrict__ mu, const float* __restrict__ lv,
                    const float* __restrict__ eps, float* __restrict__ z) {
  size_t total = (size_t)NN * LAT;
  for (size_t t = gtid(); t < total; t += GSTRIDE)
    z[t] = mu[t] + eps[t] * expf(0.5f * lv[t]);
}

__global__ void k_wnorm(const float* __restrict__ w, float* __restrict__ wn) {
  __shared__ float s[LAT];
  int i = threadIdx.x;
  s[i] = w[i] * w[i];
  __syncthreads();
  for (int st = LAT / 2; st > 0; st >>= 1) {
    if (i < st) s[i] += s[i + st];
    __syncthreads();
  }
  if (i == 0) wn[0] = sqrtf(s[0]);
}

__global__ void k_score(const float* __restrict__ z, const float* __restrict__ w,
                        const float* __restrict__ wn, float* __restrict__ sc) {
  size_t t = gtid();
  if (t >= (size_t)NN) return;
  const float* zp = z + t * LAT;
  float acc = 0.0f;
  for (int k = 0; k < LAT; ++k) acc += zp[k] * w[k];
  sc[t] = tanhf(acc / wn[0]);
}

// exact rank-based top-k per graph; matches jax.lax.top_k order (desc, stable ties)
__global__ void k_topk(const float* __restrict__ score, int* __restrict__ perm) {
  int g = blockIdx.x;
  int i = threadIdx.x;               // 1024 threads
  __shared__ float s[NPER];
  s[i] = score[(size_t)g * NPER + i];
  __syncthreads();
  float si = s[i];
  int rank = 0;
  for (int j = 0; j < NPER; ++j) {
    float sj = s[j];
    rank += (sj > si) || (sj == si && j < i);
  }
  if (rank < KPER) perm[g * KPER + rank] = g * NPER + i;
}

__global__ void k_gather(const int* __restrict__ perm, const float* __restrict__ z,
                         const float* __restrict__ mu, const float* __restrict__ lv,
                         const float* __restrict__ score, float* __restrict__ out) {
  size_t total = (size_t)POOL * LAT;
  for (size_t t = gtid(); t < total; t += GSTRIDE) {
    size_t m = t / LAT;
    int c = (int)(t % LAT);
    int p = perm[m];
    size_t off = (size_t)p * LAT + c;
    out[t]                       = z[off] * score[p];
    out[(size_t)POOL * LAT + t]  = mu[off];
    out[2ull * POOL * LAT + t]   = lv[off];
  }
}

__global__ void k_meta(const int* __restrict__ perm, float* __restrict__ out) {
  for (size_t t = gtid(); t < (size_t)POOL; t += GSTRIDE) {
    int p = perm[t];
    out[3ull * POOL * LAT + t]        = (float)(p / NPER);   // batch_pool
    out[3ull * POOL * LAT + POOL + t] = (float)p;            // perm
  }
}

extern "C" void kernel_launch(void* const* d_in, const int* in_sizes, int n_in,
                              void* d_out, int out_size, void* d_ws, size_t ws_size,
                              hipStream_t stream) {
  const float* xraw = (const float*)d_in[0];
  const float* pe   = (const float*)d_in[1];
  const int*   ei   = (const int*)d_in[2];
  const float* eps  = (const float*)d_in[4];
  const float* W1   = (const float*)d_in[5];
  const float* b1   = (const float*)d_in[6];
  const float* Wmu  = (const float*)d_in[7];
  const float* bmu  = (const float*)d_in[8];
  const float* Wlv  = (const float*)d_in[9];
  const float* blv  = (const float*)d_in[10];
  const float* wplp = (const float*)d_in[11];
  float* out = (float*)d_out;

  const int* src = ei;
  const int* dst = ei + EE;

  // workspace carving (all sizes multiples of 4 floats -> 16B alignment kept)
  float* base  = (float*)d_ws;
  size_t o = 0;
  float* dinv  = base + o; o += NN;
  float* score = base + o; o += NN;
  int*   perm  = (int*)(base + o); o += POOL;
  float* wn    = base + o; o += 16;
  float* bufA  = base + o; o += (size_t)NN * HID;   // XW, later mu0|lv0
  float* bufB  = base + o; o += (size_t)NN * HID;   // H agg -> h -> z
  float* bufC  = base + o; o += (size_t)NN * LAT;   // mu agg -> mu
  float* bufD  = base + o; o += (size_t)NN * LAT;   // lv agg -> lv
  float* mu0 = bufA;
  float* lv0 = bufA + (size_t)NN * LAT;

  const int T = 256;

  // init (must run every call: graph replay determinism)
  k_fill<<<NN / T, T, 0, stream>>>(dinv, 1.0f, (size_t)NN);          // self-loop degree
  k_fill<<<8192, T, 0, stream>>>(bufB, 0.0f, (size_t)NN * HID);
  k_fill<<<4096, T, 0, stream>>>(bufC, 0.0f, (size_t)NN * LAT);
  k_fill<<<4096, T, 0, stream>>>(bufD, 0.0f, (size_t)NN * LAT);

  // degree + D^-1/2
  k_deg<<<4096, T, 0, stream>>>(dst, dinv);
  k_rsqrt<<<NN / T, T, 0, stream>>>(dinv);

  // layer 1: XW = [x_raw|pe] @ W1 (WMMA, LDS-staged A), scatter, finish(+relu)
  k_gemm1<<<(NN / 16) * 2, T, 0, stream>>>(xraw, pe, W1, bufA);
  k_prop<<<8192, T, 0, stream>>>(src, dst, dinv, bufA, bufB, HID);
  k_finish<<<8192, T, 0, stream>>>(bufB, bufA, dinv, b1, HID, 1);    // bufB = h

  // layers mu / logvar (WMMA)
  k_gemm_h<<<NN / 16, T, 0, stream>>>(bufB, Wmu, mu0, LAT);
  k_gemm_h<<<NN / 16, T, 0, stream>>>(bufB, Wlv, lv0, LAT);
  k_prop<<<8192, T, 0, stream>>>(src, dst, dinv, mu0, bufC, LAT);
  k_prop<<<8192, T, 0, stream>>>(src, dst, dinv, lv0, bufD, LAT);
  k_finish<<<4096, T, 0, stream>>>(bufC, mu0, dinv, bmu, LAT, 0);    // bufC = mu
  k_finish<<<4096, T, 0, stream>>>(bufD, lv0, dinv, blv, LAT, 0);    // bufD = logvar

  // z, score, top-k, gather
  k_z<<<8192, T, 0, stream>>>(bufC, bufD, eps, bufB);                // bufB = z (h dead)
  k_wnorm<<<1, LAT, 0, stream>>>(wplp, wn);
  k_score<<<NN / T, T, 0, stream>>>(bufB, wplp, wn, score);
  k_topk<<<NGR, NPER, 0, stream>>>(score, perm);
  k_gather<<<4096, T, 0, stream>>>(perm, bufB, bufC, bufD, score, out);
  k_meta<<<POOL / T, T, 0, stream>>>(perm, out);
}
